// GatedSSMLayer_62302795596637
// MI455X (gfx1250) — compile-verified, matrix operand order checked by reading, not run on hardware
//
#include <hip/hip_runtime.h>
#include <stdint.h>
#include <math.h>

typedef __bf16 bf16_t;
typedef __attribute__((ext_vector_type(16))) __bf16 v16bf;
typedef __attribute__((ext_vector_type(8)))  __bf16 bf16x8;
typedef __attribute__((ext_vector_type(8)))  float   v8f;

#define DMODEL 1024
#define NQKVG  4096
#define BT     8192   /* B*T */
#define TSEQ   2048
#define NHEADS 16

// B-tile staging in LDS: 128 rows x 32 K bf16, padded row stride of 40 elems
// (80 B = 20 dwords; 20*lm mod 64 hits 16 distinct banks -> conflict-free).
// Triple buffered: one barrier per K-step is sufficient (write at iter kt
// targets the buffer last read at kt-2, ordered by barrier(kt-1)).
#define KSTEP    32
#define ROWPAD   40
#define BUFELEMS (128 * ROWPAD)
#define NBUF     3
#define NKT      (DMODEL / KSTEP)

// ---------------------------------------------------------------------------
// Weight/bias prep: fp32 -> bf16, concat [WQ;WK;WV;Wg] -> Wcat (4096x1024),
// WO -> WOb (1024x1024), biases concat -> bcat (4096 f32).
// ---------------------------------------------------------------------------
__global__ __launch_bounds__(256) void prep_kernel(
    const float* __restrict__ WQ, const float* __restrict__ WK,
    const float* __restrict__ WV, const float* __restrict__ Wg,
    const float* __restrict__ WO,
    const float* __restrict__ bQ, const float* __restrict__ bK,
    const float* __restrict__ bV, const float* __restrict__ bg,
    bf16_t* __restrict__ Wcat, bf16_t* __restrict__ WOb,
    float* __restrict__ bcat) {
  size_t idx = (size_t)blockIdx.x * 256 + threadIdx.x;
  if (idx < (size_t)NQKVG * DMODEL) {
    int row = (int)(idx >> 10);
    int col = (int)(idx & 1023);
    int grp = row >> 10;
    int r   = row & 1023;
    const float* src = (grp == 0) ? WQ : (grp == 1) ? WK : (grp == 2) ? WV : Wg;
    Wcat[idx] = (bf16_t)src[(size_t)r * DMODEL + col];
  } else if (idx < (size_t)NQKVG * DMODEL + (size_t)DMODEL * DMODEL) {
    size_t i2 = idx - (size_t)NQKVG * DMODEL;
    WOb[i2] = (bf16_t)WO[i2];
  } else if (idx < (size_t)NQKVG * DMODEL + (size_t)DMODEL * DMODEL + NQKVG) {
    int n = (int)(idx - ((size_t)NQKVG * DMODEL + (size_t)DMODEL * DMODEL));
    const float* b = (n < 1024) ? bQ : (n < 2048) ? bK : (n < 3072) ? bV : bg;
    bcat[n] = b[n & 1023];
  }
}

// ---------------------------------------------------------------------------
// LayerNorm: one block per row of x (8192 rows x 1024), bf16 output.
// ---------------------------------------------------------------------------
__global__ __launch_bounds__(256) void ln_kernel(
    const float* __restrict__ x, const float* __restrict__ gamma,
    const float* __restrict__ beta, bf16_t* __restrict__ xn) {
  int row = blockIdx.x;
  int tid = threadIdx.x;
  const float* xr = x + (size_t)row * DMODEL;
  float4 v = *(const float4*)(xr + tid * 4);
  float s  = v.x + v.y + v.z + v.w;
  float s2 = v.x * v.x + v.y * v.y + v.z * v.z + v.w * v.w;
  __shared__ float r0[256], r1[256];
  r0[tid] = s; r1[tid] = s2;
  __syncthreads();
  for (int off = 128; off > 0; off >>= 1) {
    if (tid < off) { r0[tid] += r0[tid + off]; r1[tid] += r1[tid + off]; }
    __syncthreads();
  }
  float mu   = r0[0] * (1.0f / 1024.0f);
  float var  = r1[0] * (1.0f / 1024.0f) - mu * mu;
  float rstd = rsqrtf(var + 1e-5f);
  float4 g = *(const float4*)(gamma + tid * 4);
  float4 b = *(const float4*)(beta  + tid * 4);
  bf16_t* o = xn + (size_t)row * DMODEL + tid * 4;
  o[0] = (bf16_t)((v.x - mu) * rstd * g.x + b.x);
  o[1] = (bf16_t)((v.y - mu) * rstd * g.y + b.y);
  o[2] = (bf16_t)((v.z - mu) * rstd * g.z + b.z);
  o[3] = (bf16_t)((v.w - mu) * rstd * g.w + b.w);
}

// ---------------------------------------------------------------------------
// WMMA fragment assemble from two 16B halves (K-contiguous storage; A/B
// fragment layouts are symmetric: lane = row/col (lane&15), K striped as
// [kb..kb+7] in elems 0..7 and [kb+16..kb+23] in elems 8..15, kb=(lane>>4)*8).
// ---------------------------------------------------------------------------
__device__ __forceinline__ v16bf make_frag(bf16x8 lo, bf16x8 hi) {
  v16bf r;
#pragma unroll
  for (int e = 0; e < 8; ++e) { r[e] = lo[e]; r[e + 8] = hi[e]; }
  return r;
}

__device__ __forceinline__ v16bf load_frag_g(const bf16_t* __restrict__ p) {
  return make_frag(*(const bf16x8*)p, *(const bf16x8*)(p + 16));
}

// ---------------------------------------------------------------------------
// Async copy of one 128x32 B tile into LDS. Each of the 256 threads issues
// two global_load_async_to_lds_b128 (2 x 16B), tracked by ASYNCcnt -- no VGPR
// round trip, overlapped with WMMA on the current buffer.
// ---------------------------------------------------------------------------
__device__ __forceinline__ void issue_b_tile(const bf16_t* __restrict__ Wk,
                                             bf16_t* ldsbuf, int tid) {
#pragma unroll
  for (int u = 0; u < 2; ++u) {
    int l   = tid + u * 256;          // 0..511 : 128 rows x 4 x 16B segments
    int row = l >> 2;
    int seg = l & 3;
    const bf16_t* g = Wk + (size_t)row * DMODEL + seg * 8;
    uint32_t lo = (uint32_t)(uintptr_t)&ldsbuf[row * ROWPAD + seg * 8];
    asm volatile("global_load_async_to_lds_b128 %0, %1, off"
                 :: "v"(lo), "v"(g) : "memory");
  }
}

// ---------------------------------------------------------------------------
// WMMA GEMM core: block = 256 thr (8 waves), tile 128(M) x 128(N); wave owns
// a 16x128 strip (8 f32 accumulators). A from global (per-wave private rows),
// B via async-LDS triple buffer; uniform loop body, one barrier per K-step.
// ---------------------------------------------------------------------------
#define GEMM_CORE(A, W, acc, btile)                                            \
  int tid  = threadIdx.x;                                                      \
  int lane = tid & 31;                                                         \
  int wave = tid >> 5;                                                         \
  int m0 = blockIdx.y * 128 + wave * 16;                                       \
  int n0 = blockIdx.x * 128;                                                   \
  int lm = lane & 15;                                                          \
  int kb = (lane >> 4) * 8;                                                    \
  const bf16_t* aptr = (A) + (size_t)(m0 + lm) * DMODEL + kb;                  \
  const bf16_t* Wn   = (W) + (size_t)n0 * DMODEL;                              \
  v8f acc[8];                                                                  \
  _Pragma("unroll")                                                            \
  for (int j = 0; j < 8; ++j) acc[j] = (v8f){0, 0, 0, 0, 0, 0, 0, 0};          \
  issue_b_tile(Wn, &btile[0], tid);                                            \
  int cur = 0;                                                                 \
  for (int kt = 0; kt < NKT; ++kt) {                                           \
    int k   = kt * KSTEP;                                                      \
    int nxt = (cur + 1 == NBUF) ? 0 : cur + 1;                                 \
    int kn  = (kt + 1 < NKT) ? k + KSTEP : 0; /* wrap: dead prefetch, uniform */\
    issue_b_tile(Wn + kn, &btile[nxt * BUFELEMS], tid);                        \
    asm volatile("s_wait_asynccnt 0x2" ::: "memory");                          \
    __syncthreads(); /* tile resident in cur; also orders buffer reuse */      \
    const bf16_t* curb = &btile[cur * BUFELEMS];                               \
    v16bf a = load_frag_g(aptr + k);                                           \
    _Pragma("unroll")                                                          \
    for (int j = 0; j < 8; ++j) {                                              \
      const bf16_t* bp = &curb[(j * 16 + lm) * ROWPAD + kb];                   \
      v16bf bfrag = make_frag(*(const bf16x8*)bp, *(const bf16x8*)(bp + 16));  \
      acc[j] = __builtin_amdgcn_wmma_f32_16x16x32_bf16(                        \
          false, a, false, bfrag, (short)0, acc[j], false, false);             \
    }                                                                          \
    cur = nxt;                                                                 \
  }

// GEMM 1: QKVG[8192,4096] = xn @ Wcat^T + bcat (bf16 out)
__global__ __launch_bounds__(256) void gemm_qkvg_kernel(
    const bf16_t* __restrict__ A, const bf16_t* __restrict__ W,
    const float* __restrict__ bias, bf16_t* __restrict__ C) {
  __shared__ bf16_t btile[NBUF * BUFELEMS];
  GEMM_CORE(A, W, acc, btile)
  int mb = m0 + (lane >> 4) * 8;   // C/D layout: VGPR r -> M=r / r+8 per half-wave
#pragma unroll
  for (int j = 0; j < 8; ++j) {
    int n = n0 + j * 16 + lm;
    float bn = bias[n];
#pragma unroll
    for (int r = 0; r < 8; ++r)
      C[(size_t)(mb + r) * NQKVG + n] = (bf16_t)(acc[j][r] + bn);
  }
}

// GEMM 2: out[8192,1024] = O @ WO^T + bO + residual(x)  (f32 out)
__global__ __launch_bounds__(256) void gemm_out_kernel(
    const bf16_t* __restrict__ A, const bf16_t* __restrict__ W,
    const float* __restrict__ bO, const float* __restrict__ xres,
    float* __restrict__ out) {
  __shared__ bf16_t btile[NBUF * BUFELEMS];
  GEMM_CORE(A, W, acc, btile)
  int mb = m0 + (lane >> 4) * 8;
#pragma unroll
  for (int j = 0; j < 8; ++j) {
    int n = n0 + j * 16 + lm;
    float bn = bO[n];
#pragma unroll
    for (int r = 0; r < 8; ++r) {
      size_t idx = (size_t)(mb + r) * DMODEL + n;
      out[idx] = acc[j][r] + bn + xres[idx];
    }
  }
}

// ---------------------------------------------------------------------------
// Recurrence: one block per (head, batch); 64x64 fp32 state in registers.
// h = alpha[i]*h + k[i]*v[j]; o[j] = sum_i q[i]*h[i][j]; o *= silu(gate).
// ---------------------------------------------------------------------------
__global__ __launch_bounds__(256) void recur_kernel(
    const bf16_t* __restrict__ QKVG, const float* __restrict__ alogit,
    bf16_t* __restrict__ O) {
  int h = blockIdx.x;            // 0..15
  int b = blockIdx.y;            // 0..3
  int tid   = threadIdx.x;
  int j     = tid & 63;
  int ibase = (tid >> 6) << 4;   // 16 state rows per thread
  float alpha[16];
#pragma unroll
  for (int ii = 0; ii < 16; ++ii)
    alpha[ii] = 1.0f / (1.0f + __expf(-alogit[h * 64 + ibase + ii]));
  float hst[16];
#pragma unroll
  for (int ii = 0; ii < 16; ++ii) hst[ii] = 0.0f;

  __shared__ float sq[64], sk[64], sv[64], sg[64], sacc[256];
  const bf16_t* base  = QKVG + (size_t)b * TSEQ * NQKVG + h * 64;
  bf16_t*       obase = O    + (size_t)b * TSEQ * DMODEL + h * 64;

  for (int t = 0; t < TSEQ; ++t) {
    const bf16_t* rp = base + (size_t)t * NQKVG;
    if (tid < 64) {
      sq[tid] = (float)rp[tid];
      sk[tid] = (float)rp[1024 + tid];
      sv[tid] = (float)rp[2048 + tid];
      sg[tid] = (float)rp[3072 + tid];
    }
    __syncthreads();
    float vj  = sv[j];
    float acc = 0.0f;
#pragma unroll
    for (int ii = 0; ii < 16; ++ii) {
      float hn = alpha[ii] * hst[ii] + sk[ibase + ii] * vj;
      hst[ii] = hn;
      acc += sq[ibase + ii] * hn;
    }
    sacc[tid] = acc;
    __syncthreads();
    if (tid < 64) {
      float o = sacc[tid] + sacc[tid + 64] + sacc[tid + 128] + sacc[tid + 192];
      float g = sg[tid];
      o *= g / (1.0f + __expf(-g));          // SiLU gate
      obase[(size_t)t * DMODEL + tid] = (bf16_t)o;
    }
  }
}

// ---------------------------------------------------------------------------
extern "C" void kernel_launch(void* const* d_in, const int* in_sizes, int n_in,
                              void* d_out, int out_size, void* d_ws, size_t ws_size,
                              hipStream_t stream) {
  (void)in_sizes; (void)n_in; (void)out_size; (void)ws_size;
  const float* x      = (const float*)d_in[0];
  const float* WQ     = (const float*)d_in[1];
  const float* bQ     = (const float*)d_in[2];
  const float* WK     = (const float*)d_in[3];
  const float* bK     = (const float*)d_in[4];
  const float* WV     = (const float*)d_in[5];
  const float* bV     = (const float*)d_in[6];
  const float* WO     = (const float*)d_in[7];
  const float* bO     = (const float*)d_in[8];
  const float* Wg     = (const float*)d_in[9];
  const float* bg     = (const float*)d_in[10];
  const float* alogit = (const float*)d_in[11];
  const float* gamma  = (const float*)d_in[12];
  const float* beta   = (const float*)d_in[13];
  float* out = (float*)d_out;

  // carve workspace (all offsets 256B aligned)
  char* ws = (char*)d_ws;
  size_t off = 0;
  auto take = [&](size_t bytes) -> void* {
    void* p = ws + off;
    off += (bytes + 255) & ~(size_t)255;
    return p;
  };
  bf16_t* XN   = (bf16_t*)take((size_t)BT * DMODEL * sizeof(bf16_t));
  bf16_t* WCAT = (bf16_t*)take((size_t)NQKVG * DMODEL * sizeof(bf16_t));
  bf16_t* WOB  = (bf16_t*)take((size_t)DMODEL * DMODEL * sizeof(bf16_t));
  float*  BCAT = (float*) take((size_t)NQKVG * sizeof(float));
  bf16_t* QKVG = (bf16_t*)take((size_t)BT * NQKVG * sizeof(bf16_t));
  bf16_t* OB   = (bf16_t*)take((size_t)BT * DMODEL * sizeof(bf16_t));

  size_t prep_elems = (size_t)NQKVG * DMODEL + (size_t)DMODEL * DMODEL + NQKVG;
  int prep_blocks = (int)((prep_elems + 255) / 256);
  prep_kernel<<<prep_blocks, 256, 0, stream>>>(WQ, WK, WV, Wg, WO, bQ, bK, bV, bg,
                                               WCAT, WOB, BCAT);
  ln_kernel<<<BT, 256, 0, stream>>>(x, gamma, beta, XN);
  gemm_qkvg_kernel<<<dim3(NQKVG / 128, BT / 128), 256, 0, stream>>>(XN, WCAT, BCAT, QKVG);
  recur_kernel<<<dim3(NHEADS, 4), 256, 0, stream>>>(QKVG, alogit, OB);
  gemm_out_kernel<<<dim3(DMODEL / 128, BT / 128), 256, 0, stream>>>(OB, WOB, bO, x, out);
}
